// GGNNModel_62912680952413
// MI455X (gfx1250) — matrix operand order repaired and checked by reading, not compiled, not run on hardware
//
#include <hip/hip_runtime.h>
#include <math.h>

// GGNN on MI455X (gfx1250, wave32). All 192-wide GEMMs use v_wmma_f32_16x16x32_f16.
// Message passing is the roofline limiter; gather side reads a f16 mirror of h
// (halves gather bytes; identical rounding to what WMMA consumes), scatter side
// stays exact f32 global atomics resolving in the 192MB L2 (h+h16+agg ~ 96MB).

typedef __attribute__((ext_vector_type(16))) _Float16 v16h;
typedef __attribute__((ext_vector_type(8)))  float    v8f;

#define CCH   96      // channels
#define C2    192     // 2*C
#define KCH   6       // 192 / 32 K-chunks per WMMA dot
#define CTIL  6       // 96 / 16 column tiles

__device__ __forceinline__ float sigmoidf_(float x) {
  return 1.0f / (1.0f + __expf(-x));
}

// ---------------------------------------------------------------------------
// Pre-swizzle a 192x96 f32 row-major weight into WMMA B-fragment order:
// frag (kk, ct): 32 lanes x 16 halves, b[i] = W[kk*32 + (lane>=16)*8 + (i&7) + (i>>3)*16][ct*16 + lane%16]
// -> each lane later loads one contiguous 32B v16h.
// ---------------------------------------------------------------------------
__global__ void swz_weight(const float* __restrict__ W, _Float16* __restrict__ dst) {
  int tid = blockIdx.x * blockDim.x + threadIdx.x;
  const int total = KCH * CTIL * 32 * 16;   // 18432
  if (tid >= total) return;
  int i    = tid & 15;
  int lane = (tid >> 4) & 31;
  int frag = tid >> 9;
  int kk = frag / CTIL;
  int ct = frag % CTIL;
  int hi = lane >> 4;
  int col = ct * 16 + (lane & 15);
  int k   = kk * 32 + hi * 8 + (i & 7) + (i >> 3) * 16;
  dst[tid] = (_Float16)W[k * CCH + col];
}

// ---------------------------------------------------------------------------
// Degree / normalization
// ---------------------------------------------------------------------------
__global__ void init_deg(float* deg, int n) {
  int i = blockIdx.x * blockDim.x + threadIdx.x;
  if (i < n) deg[i] = 1.0f;                       // self loop
}
__global__ void deg_edges(const long long* __restrict__ col, float* deg, int e) {
  int i = blockIdx.x * blockDim.x + threadIdx.x;
  if (i < e) atomicAdd(&deg[(int)col[i]], 1.0f);
}
__global__ void make_dis(float* d, int n) {
  int i = blockIdx.x * blockDim.x + threadIdx.x;
  if (i < n) d[i] = rsqrtf(d[i]);                 // deg >= 1 always
}

// ---------------------------------------------------------------------------
// h0 = relu(relu(x @ Wp + bp) @ Wi + bi)   (N x 16 -> 32 -> 96), tiny VALU work
// Writes f32 h0 and its f16 mirror (gather side of layer 1).
// ---------------------------------------------------------------------------
__global__ __launch_bounds__(96) void h0_kernel(const float* __restrict__ x,
    const float* __restrict__ Wp, const float* __restrict__ bp,
    const float* __restrict__ Wi, const float* __restrict__ bi,
    float* __restrict__ h0, _Float16* __restrict__ h0h) {
  __shared__ float xv[16];
  __shared__ float p[32];
  int n = blockIdx.x, t = threadIdx.x;
  if (t < 16) xv[t] = x[n * 16 + t];
  __syncthreads();
  if (t < 32) {
    float s = bp[t];
#pragma unroll
    for (int k = 0; k < 16; ++k) s += xv[k] * Wp[k * 32 + t];
    p[t] = fmaxf(s, 0.0f);
  }
  __syncthreads();
  float s = bi[t];
#pragma unroll
  for (int k = 0; k < 32; ++k) s += p[k] * Wi[k * CCH + t];
  float v = fmaxf(s, 0.0f);
  h0[n * CCH + t]  = v;
  h0h[n * CCH + t] = (_Float16)v;
}

// ---------------------------------------------------------------------------
// agg init (self-loop term) + edge scatter with inline symmetric norm.
// Gather reads the f16 mirror of h; accumulation is exact f32 atomics.
// ---------------------------------------------------------------------------
__global__ void agg_init(const _Float16* __restrict__ h16, const float* __restrict__ dis,
                         float* __restrict__ agg, int n) {
  int tid = blockIdx.x * blockDim.x + threadIdx.x;
  if (tid >= n * CCH) return;
  int node = tid / CCH;
  float sn = dis[node];
  agg[tid] = sn * sn * (float)h16[tid];
}

__global__ void agg_edges(const long long* __restrict__ row,
                          const long long* __restrict__ col,
                          const _Float16* __restrict__ h16, const float* __restrict__ dis,
                          float* __restrict__ agg, int e) {
  // 2 channels per thread: 4B coalesced gather loads, 2 f32 atomics.
  long long tid = (long long)blockIdx.x * blockDim.x + threadIdx.x;
  if (tid >= (long long)e * (CCH / 2)) return;
  int eidx = (int)(tid / (CCH / 2));
  int c    = 2 * (int)(tid % (CCH / 2));
  int r  = (int)row[eidx];
  int cl = (int)col[eidx];
  float w = dis[r] * dis[cl];
  const _Float16* hp = &h16[(long long)r * CCH + c];
  float v0 = w * (float)hp[0];
  float v1 = w * (float)hp[1];
  float* ap = &agg[(long long)cl * CCH + c];
  atomicAdd(ap,     v0);
  atomicAdd(ap + 1, v1);
}

// ---------------------------------------------------------------------------
// Tied GGNN layer: one 16-row tile per block, 6 waves = 6 column tiles of 16.
//   xa = f16([h | agg])  in LDS, A-fragments assembled per ISA 16-bit A layout.
//   z/r via WMMA -> barrier -> overwrite h-half of xa with f16(r*h) -> barrier
//   -> hc via WMMA -> gated in-place update (block only touches its own rows).
// Writes f32 h and its f16 mirror (gather side of next layer).
// ---------------------------------------------------------------------------
__global__ __launch_bounds__(192) void layer_gemm(
    const float* __restrict__ hin, const float* __restrict__ agg,
    const _Float16* __restrict__ WzS, const _Float16* __restrict__ WrS,
    const _Float16* __restrict__ WhS,
    const float* __restrict__ bz, const float* __restrict__ br,
    const float* __restrict__ bh,
    float* __restrict__ hout, _Float16* __restrict__ hout16, int nNodes) {
  __shared__ _Float16 xa[16][C2];
  __shared__ float    hf[16][CCH];
  int tid = threadIdx.x;
  int m0  = blockIdx.x * 16;

#pragma unroll
  for (int it = 0; it < 16; ++it) {
    int idx = it * 192 + tid;
    int r = idx / C2, c = idx % C2;
    int gr = m0 + r;
    float v = 0.0f;
    if (gr < nNodes)
      v = (c < CCH) ? hin[(long long)gr * CCH + c]
                    : agg[(long long)gr * CCH + (c - CCH)];
    xa[r][c] = (_Float16)v;
    if (c < CCH) hf[r][c] = v;
  }
  __syncthreads();

  int wv = tid >> 5, lane = tid & 31;
  int m = lane & 15, hi = lane >> 4;

  v8f az = {}; v8f ar = {};
  const v16h* Bz = (const v16h*)WzS;
  const v16h* Br = (const v16h*)WrS;
  const v16h* Bh = (const v16h*)WhS;
#pragma unroll
  for (int kk = 0; kk < KCH; ++kk) {
    v16h a;
    const _Float16* xr = &xa[m][kk * 32 + hi * 8];
#pragma unroll
    for (int i = 0; i < 8; ++i) { a[i] = xr[i]; a[8 + i] = xr[16 + i]; }
    int bi_ = (kk * CTIL + wv) * 32 + lane;
    az = __builtin_amdgcn_wmma_f32_16x16x32_f16(false, a, false, Bz[bi_], (short)0, az, false, false);
    ar = __builtin_amdgcn_wmma_f32_16x16x32_f16(false, a, false, Br[bi_], (short)0, ar, false, false);
  }

  int col = wv * 16 + m;
  float bzv = bz[col], brv = br[col];
  float zf[8], hold[8];
  __syncthreads();                       // everyone done reading xa for z/r
#pragma unroll
  for (int r8 = 0; r8 < 8; ++r8) {
    int row = r8 + hi * 8;
    float zz = sigmoidf_(az[r8] + bzv);
    float rr = sigmoidf_(ar[r8] + brv);
    float hv = hf[row][col];
    zf[r8] = zz; hold[r8] = hv;
    xa[row][col] = (_Float16)(rr * hv);  // [r*h | agg] now lives in xa
  }
  __syncthreads();

  v8f ah = {};
#pragma unroll
  for (int kk = 0; kk < KCH; ++kk) {
    v16h a;
    const _Float16* xr = &xa[m][kk * 32 + hi * 8];
#pragma unroll
    for (int i = 0; i < 8; ++i) { a[i] = xr[i]; a[8 + i] = xr[16 + i]; }
    int bi_ = (kk * CTIL + wv) * 32 + lane;
    ah = __builtin_amdgcn_wmma_f32_16x16x32_f16(false, a, false, Bh[bi_], (short)0, ah, false, false);
  }

  float bhv = bh[col];
#pragma unroll
  for (int r8 = 0; r8 < 8; ++r8) {
    int row = r8 + hi * 8;
    int gr = m0 + row;
    float hc = fmaxf(ah[r8] + bhv, 0.0f);
    float hn = (1.0f - zf[r8]) * hold[r8] + zf[r8] * hc;
    if (gr < nNodes) {
      hout[(long long)gr * CCH + col]   = hn;
      hout16[(long long)gr * CCH + col] = (_Float16)hn;
    }
  }
}

// ---------------------------------------------------------------------------
// Final head: relu([h0|h] @ W1 + b1) via WMMA, then 96->2 sigmoid.
// ---------------------------------------------------------------------------
__global__ __launch_bounds__(192) void final_kernel(
    const float* __restrict__ h0, const float* __restrict__ h,
    const _Float16* __restrict__ W1S, const float* __restrict__ b1,
    const float* __restrict__ W2, const float* __restrict__ b2,
    float* __restrict__ out, int nNodes) {
  __shared__ _Float16 xa[16][C2];
  __shared__ float    o1[16][CCH];
  int tid = threadIdx.x;
  int m0  = blockIdx.x * 16;

#pragma unroll
  for (int it = 0; it < 16; ++it) {
    int idx = it * 192 + tid;
    int r = idx / C2, c = idx % C2;
    int gr = m0 + r;
    float v = 0.0f;
    if (gr < nNodes)
      v = (c < CCH) ? h0[(long long)gr * CCH + c]
                    : h[(long long)gr * CCH + (c - CCH)];
    xa[r][c] = (_Float16)v;
  }
  __syncthreads();

  int wv = tid >> 5, lane = tid & 31;
  int m = lane & 15, hi = lane >> 4;
  const v16h* B1 = (const v16h*)W1S;
  v8f acc = {};
#pragma unroll
  for (int kk = 0; kk < KCH; ++kk) {
    v16h a;
    const _Float16* xr = &xa[m][kk * 32 + hi * 8];
#pragma unroll
    for (int i = 0; i < 8; ++i) { a[i] = xr[i]; a[8 + i] = xr[16 + i]; }
    int bi_ = (kk * CTIL + wv) * 32 + lane;
    acc = __builtin_amdgcn_wmma_f32_16x16x32_f16(false, a, false, B1[bi_], (short)0, acc, false, false);
  }
  int col = wv * 16 + m;
  float b1v = b1[col];
#pragma unroll
  for (int r8 = 0; r8 < 8; ++r8)
    o1[r8 + hi * 8][col] = fmaxf(acc[r8] + b1v, 0.0f);
  __syncthreads();

  if (tid < 32) {
    int row = tid & 15;
    int j   = tid >> 4;
    int gr = m0 + row;
    float s = b2[j];
#pragma unroll
    for (int k = 0; k < CCH; ++k) s += o1[row][k] * W2[k * 2 + j];
    if (gr < nNodes) out[(long long)gr * 2 + j] = sigmoidf_(s);
  }
}

// ---------------------------------------------------------------------------
extern "C" void kernel_launch(void* const* d_in, const int* in_sizes, int n_in,
                              void* d_out, int out_size, void* d_ws, size_t ws_size,
                              hipStream_t stream) {
  const float*     x  = (const float*)d_in[0];
  const long long* ei = (const long long*)d_in[1];
  const float* Wp = (const float*)d_in[2];
  const float* bp = (const float*)d_in[3];
  const float* Wi = (const float*)d_in[4];
  const float* bi = (const float*)d_in[5];
  const float* Wz = (const float*)d_in[6];
  const float* bz = (const float*)d_in[7];
  const float* Wr = (const float*)d_in[8];
  const float* br = (const float*)d_in[9];
  const float* Wh = (const float*)d_in[10];
  const float* bh = (const float*)d_in[11];
  const float* W1 = (const float*)d_in[12];
  const float* b1 = (const float*)d_in[13];
  const float* W2 = (const float*)d_in[14];
  const float* b2 = (const float*)d_in[15];

  const int N = in_sizes[0] / 16;
  const int E = in_sizes[1] / 2;

  char* ws = (char*)d_ws;
  size_t off = 0;
  auto take = [&](size_t bytes) -> void* {
    void* p = ws + off;
    off = (off + bytes + 255) & ~(size_t)255;
    return p;
  };
  const size_t swzBytes = (size_t)KCH * CTIL * 32 * 16 * sizeof(_Float16);
  _Float16* WzS = (_Float16*)take(swzBytes);
  _Float16* WrS = (_Float16*)take(swzBytes);
  _Float16* WhS = (_Float16*)take(swzBytes);
  _Float16* W1S = (_Float16*)take(swzBytes);
  float*     dis = (float*)take((size_t)N * sizeof(float));
  float*     h0  = (float*)take((size_t)N * CCH * sizeof(float));
  float*     h   = (float*)take((size_t)N * CCH * sizeof(float));
  float*     agg = (float*)take((size_t)N * CCH * sizeof(float));
  _Float16*  h0h = (_Float16*)take((size_t)N * CCH * sizeof(_Float16));
  _Float16*  h16 = (_Float16*)take((size_t)N * CCH * sizeof(_Float16));
  (void)ws_size; (void)n_in; (void)out_size;

  const int swzTot = KCH * CTIL * 32 * 16;
  swz_weight<<<(swzTot + 255) / 256, 256, 0, stream>>>(Wz, WzS);
  swz_weight<<<(swzTot + 255) / 256, 256, 0, stream>>>(Wr, WrS);
  swz_weight<<<(swzTot + 255) / 256, 256, 0, stream>>>(Wh, WhS);
  swz_weight<<<(swzTot + 255) / 256, 256, 0, stream>>>(W1, W1S);

  init_deg<<<(N + 255) / 256, 256, 0, stream>>>(dis, N);
  deg_edges<<<(E + 255) / 256, 256, 0, stream>>>(ei + E, dis, E);
  make_dis<<<(N + 255) / 256, 256, 0, stream>>>(dis, N);

  h0_kernel<<<N, 96, 0, stream>>>(x, Wp, bp, Wi, bi, h0, h0h);

  const int nTiles = (N + 15) / 16;
  const float*    hin   = h0;            // layer 1 reads h0/h0h, writes h/h16;
  const _Float16* h16in = h0h;           // later layers update h/h16 in place
  for (int l = 0; l < 6; ++l) {
    agg_init<<<((long long)N * CCH + 255) / 256, 256, 0, stream>>>(h16in, dis, agg, N);
    long long tot = (long long)E * (CCH / 2);
    agg_edges<<<(int)((tot + 255) / 256), 256, 0, stream>>>(ei, ei + E, h16in, dis, agg, E);
    layer_gemm<<<nTiles, 192, 0, stream>>>(hin, agg, WzS, WrS, WhS, bz, br, bh, h, h16, N);
    hin = h; h16in = h16;
  }

  final_kernel<<<nTiles, 192, 0, stream>>>(h0, h, W1S, b1, W2, b2, (float*)d_out, N);
}
